// RecurrentTorso_46969762349629
// MI455X (gfx1250) — matrix operand order repaired and checked by reading, compiled, not run on hardware
//
#include <hip/hip_runtime.h>
#include <math.h>

// ---------------- Problem constants (from the reference) ----------------
#define VV  32000
#define EE  512
#define HH1 2048
#define HH2 1024
#define HHL 1024      // LSTM hidden
#define BB  64
#define TT  256
#define NTOK (BB*TT)  // 16384
#define GN  (4*HHL)   // 4096 gate columns

typedef __attribute__((ext_vector_type(16))) __bf16          v16bf;
typedef __attribute__((ext_vector_type(8)))  float           v8f;
typedef __attribute__((ext_vector_type(8)))  unsigned short  ushort8;

static __device__ __forceinline__ unsigned short f32_to_bf16(float f) {
  union { float f; unsigned u; } v; v.f = f;
  unsigned u = v.u;
  unsigned r = u + 0x7fffu + ((u >> 16) & 1u);   // round-to-nearest-even
  return (unsigned short)(r >> 16);
}

static __device__ __forceinline__ v16bf make_frag(ushort8 lo, ushort8 hi) {
  union { ushort8 h[2]; v16bf v; } u;
  u.h[0] = lo; u.h[1] = hi;
  return u.v;
}

// ---------------- fp32 -> bf16 convert + transpose ----------------
// in:  [K][N] fp32 row-major;  out: [N][K] bf16 row-major. 32x32 LDS tiles.
__global__ __launch_bounds__(256)
void cvt_transpose_kernel(const float* __restrict__ in,
                          unsigned short* __restrict__ out, int K, int N) {
  __shared__ float tile[32][33];
  const int kb = blockIdx.y * 32, nb = blockIdx.x * 32;
  const int tx = threadIdx.x & 31, ty = threadIdx.x >> 5; // ty: 0..7
#pragma unroll
  for (int r = ty; r < 32; r += 8)
    tile[r][tx] = in[(size_t)(kb + r) * N + nb + tx];
  __syncthreads();
#pragma unroll
  for (int r = ty; r < 32; r += 8)
    out[(size_t)(nb + r) * K + kb + tx] = f32_to_bf16(tile[tx][r]);
}

// ---------------- Embedding gather + bf16 cast ----------------
__global__ void embed_gather_kernel(const int* __restrict__ x,
                                    const float* __restrict__ embed,
                                    unsigned short* __restrict__ Xemb) {
  int token = blockIdx.x;
  int row = x[token];
  const float* src = embed + (size_t)row * EE;
  unsigned short* dst = Xemb + (size_t)token * EE;
  for (int e = threadIdx.x; e < EE; e += 256)
    dst[e] = f32_to_bf16(src[e]);
}

// ---------------- Generic bf16 WMMA GEMM (B pre-transposed) ----------------
// C[M,N] = A[M,K](bf16, row-major) * Bt[N][ldB](bf16, weights transposed)
// (+ bias[N]) (+ accinit[M,N]); optional ReLU / bf16 out / row remap.
// 256 threads = 8 waves; macro tile BM x 128; BK = 32; double-buffered LDS.
template<int BM, bool RELU, bool OUT_BF16, bool REMAP, bool ACCINIT>
__global__ __launch_bounds__(256)
void gemm_bf16_kernel(const unsigned short* __restrict__ A,
                      const unsigned short* __restrict__ Bt,
                      const float* __restrict__ bias,
                      const float* __restrict__ accinit,
                      void* __restrict__ Cout,
                      int M, int N, int K, int ldB) {
  constexpr int BN = 128;
  constexpr int BK = 32;
  constexpr int LDP = BK + 8;           // padded LDS row (shorts), 16B-aligned
  constexpr int MT = BM / 16;
  constexpr int NT = BN / 16;
  constexpr int TILES = MT * NT;
  constexpr int TPW = TILES / 8;        // tiles per wave
  constexpr int AC = (BM * 4) / 256;    // 16B A-chunks per thread (1 or 2)
  constexpr int BC = (BN * 4) / 256;    // 16B B-chunks per thread (2)

  __shared__ alignas(16) unsigned short Alds[2][BM][LDP];
  __shared__ alignas(16) unsigned short Blds[2][BN][LDP];

  const int tid  = threadIdx.x;
  const int lane = tid & 31;
  const int w    = tid >> 5;
  const int hi   = lane >> 4;
  const int lm   = lane & 15;

  const int mBlock = blockIdx.y * BM;
  const int nBlock = blockIdx.x * BN;

  const int tm  = (w * TPW) / NT;       // constant per wave -> A frag reuse
  const int tn0 = (w * TPW) % NT;

  v8f acc[TPW];
  if (ACCINIT) {
#pragma unroll
    for (int i = 0; i < TPW; ++i)
#pragma unroll
      for (int r = 0; r < 8; ++r) {
        int row = mBlock + tm * 16 + r + 8 * hi;
        int col = nBlock + (tn0 + i) * 16 + lm;
        acc[i][r] = accinit[(size_t)row * N + col];
      }
  } else {
#pragma unroll
    for (int i = 0; i < TPW; ++i)
#pragma unroll
      for (int r = 0; r < 8; ++r) acc[i][r] = 0.0f;
  }

  ushort8 areg[AC], breg[BC];

  // ---- prologue: global chunk 0 -> regs -> LDS buf 0 ----
#pragma unroll
  for (int c = 0; c < AC; ++c) {
    int idx = tid + c * 256, row = idx >> 2, kc = (idx & 3) * 8;
    areg[c] = *(const ushort8*)(A + (size_t)(mBlock + row) * K + kc);
  }
#pragma unroll
  for (int c = 0; c < BC; ++c) {
    int idx = tid + c * 256, row = idx >> 2, kc = (idx & 3) * 8;
    breg[c] = *(const ushort8*)(Bt + (size_t)(nBlock + row) * ldB + kc);
  }
#pragma unroll
  for (int c = 0; c < AC; ++c) {
    int idx = tid + c * 256, row = idx >> 2, kc = (idx & 3) * 8;
    *(ushort8*)(&Alds[0][row][kc]) = areg[c];
  }
#pragma unroll
  for (int c = 0; c < BC; ++c) {
    int idx = tid + c * 256, row = idx >> 2, kc = (idx & 3) * 8;
    *(ushort8*)(&Blds[0][row][kc]) = breg[c];
  }

  const int nk = K / BK;
  for (int ch = 0; ch < nk; ++ch) {
    const int cur = ch & 1;
    __syncthreads();

    // ---- issue next chunk's global loads (overlap with WMMA below) ----
    if (ch + 1 < nk) {
      const int k0n = (ch + 1) * BK;
#pragma unroll
      for (int c = 0; c < AC; ++c) {
        int idx = tid + c * 256, row = idx >> 2, kc = (idx & 3) * 8;
        areg[c] = *(const ushort8*)(A + (size_t)(mBlock + row) * K + k0n + kc);
      }
#pragma unroll
      for (int c = 0; c < BC; ++c) {
        int idx = tid + c * 256, row = idx >> 2, kc = (idx & 3) * 8;
        breg[c] = *(const ushort8*)(Bt + (size_t)(nBlock + row) * ldB + k0n + kc);
      }
      if (ch + 2 < nk) {   // pull K+2 weight chunk toward the WGP
        int row = tid >> 2, kc = (tid & 3) * 8;
        __builtin_prefetch(Bt + (size_t)(nBlock + row) * ldB + (ch + 2) * BK + kc, 0, 3);
      }
    }

    // ---- compute from LDS buf `cur` ----
    ushort8 a0 = *(const ushort8*)(&Alds[cur][tm * 16 + lm][hi * 16]);
    ushort8 a1 = *(const ushort8*)(&Alds[cur][tm * 16 + lm][hi * 16 + 8]);
    v16bf afrag = make_frag(a0, a1);

    v16bf bnext;
    {
      const int n = tn0 * 16 + lm;
      bnext = make_frag(*(const ushort8*)(&Blds[cur][n][hi * 16]),
                        *(const ushort8*)(&Blds[cur][n][hi * 16 + 8]));
    }
#pragma unroll
    for (int i = 0; i < TPW; ++i) {
      v16bf bcur = bnext;
      if (i + 1 < TPW) {   // load frag i+1 before consuming frag i
        const int n = (tn0 + i + 1) * 16 + lm;
        bnext = make_frag(*(const ushort8*)(&Blds[cur][n][hi * 16]),
                          *(const ushort8*)(&Blds[cur][n][hi * 16 + 8]));
      }
      acc[i] = __builtin_amdgcn_wmma_f32_16x16x32_bf16(
          false, afrag, false, bcur, (short)0, acc[i], false, false);
    }

    // ---- stage next chunk into the other LDS buffer ----
    if (ch + 1 < nk) {
      const int nxt = cur ^ 1;
#pragma unroll
      for (int c = 0; c < AC; ++c) {
        int idx = tid + c * 256, row = idx >> 2, kc = (idx & 3) * 8;
        *(ushort8*)(&Alds[nxt][row][kc]) = areg[c];
      }
#pragma unroll
      for (int c = 0; c < BC; ++c) {
        int idx = tid + c * 256, row = idx >> 2, kc = (idx & 3) * 8;
        *(ushort8*)(&Blds[nxt][row][kc]) = breg[c];
      }
    }
  }

  // ---- store ----
#pragma unroll
  for (int i = 0; i < TPW; ++i) {
    const int col = nBlock + (tn0 + i) * 16 + lm;
    const float bv = bias ? bias[col] : 0.0f;
#pragma unroll
    for (int r = 0; r < 8; ++r) {
      int row = mBlock + tm * 16 + r + 8 * hi;
      float vv = acc[i][r] + bv;
      if (RELU) vv = vv > 0.0f ? vv : 0.0f;
      int orow = REMAP ? ((row % TT) * BB + (row / TT)) : row;
      if (OUT_BF16)
        ((unsigned short*)Cout)[(size_t)orow * N + col] = f32_to_bf16(vv);
      else
        ((float*)Cout)[(size_t)orow * N + col] = vv;
    }
  }
}

// ---------------- LSTM state init ----------------
__global__ void lstm_init_kernel(float* __restrict__ C,
                                 unsigned short* __restrict__ Hbf) {
  int i = blockIdx.x * 256 + threadIdx.x;   // 65536 elems
  C[i] = 0.0f;
  Hbf[i] = 0;
}

// ---------------- LSTM elementwise combine ----------------
__global__ void lstm_combine_kernel(const float* __restrict__ GATES,
                                    float* __restrict__ C,
                                    unsigned short* __restrict__ Hbf,
                                    float* __restrict__ hout, int last) {
  int idx = blockIdx.x * 256 + threadIdx.x;  // 64*1024
  int b = idx >> 10;
  int j = idx & 1023;
  const float* g = GATES + (size_t)b * GN;
  float iv = g[j];
  float gv = g[HHL + j];
  float fv = g[2 * HHL + j];
  float ov = g[3 * HHL + j];
  float fg = 1.0f / (1.0f + expf(-(fv + 1.0f)));
  float ig = 1.0f / (1.0f + expf(-iv));
  float og = 1.0f / (1.0f + expf(-ov));
  float c = fg * C[idx] + ig * tanhf(gv);
  float h = og * tanhf(c);
  C[idx] = c;
  Hbf[idx] = f32_to_bf16(h);
  if (last) hout[idx] = h;
}

// ---------------- Host launch ----------------
extern "C" void kernel_launch(void* const* d_in, const int* in_sizes, int n_in,
                              void* d_out, int out_size, void* d_ws, size_t ws_size,
                              hipStream_t stream) {
  (void)in_sizes; (void)n_in; (void)out_size; (void)ws_size;
  const int*   x     = (const int*)  d_in[0];
  const float* embed = (const float*)d_in[1];
  const float* W1    = (const float*)d_in[2];
  const float* b1    = (const float*)d_in[3];
  const float* W2    = (const float*)d_in[4];
  const float* b2    = (const float*)d_in[5];
  const float* Wl    = (const float*)d_in[6];
  const float* bl    = (const float*)d_in[7];
  float* out = (float*)d_out;

  // ---- workspace carve-up (256B aligned) ----
  char* ws = (char*)d_ws;
  size_t off = 0;
  auto alloc = [&](size_t bytes) -> char* {
    char* p = ws + off;
    off = (off + bytes + 255) & ~(size_t)255;
    return p;
  };
  unsigned short* W1t  = (unsigned short*)alloc((size_t)HH1 * EE  * 2); // [2048][512]
  unsigned short* W2t  = (unsigned short*)alloc((size_t)HH2 * HH1 * 2); // [1024][2048]
  unsigned short* Wlt  = (unsigned short*)alloc((size_t)GN * (HH2+HHL) * 2); // [4096][2048]
  unsigned short* Xemb = (unsigned short*)alloc((size_t)NTOK * EE  * 2);
  unsigned short* H1a  = (unsigned short*)alloc((size_t)NTOK * HH1 * 2);
  unsigned short* Hmlp = (unsigned short*)alloc((size_t)NTOK * HH2 * 2);
  float*          GX   = (float*)         alloc((size_t)NTOK * GN  * 4); // [t][b][4096]
  float*          GAT  = (float*)         alloc((size_t)BB   * GN  * 4);
  float*          Cst  = (float*)         alloc((size_t)BB   * HHL * 4);
  unsigned short* Hbf  = (unsigned short*)alloc((size_t)BB   * HHL * 2);

  // Wlt rows are gate columns; k 0..1023 = x part, k 1024..2047 = h part
  const unsigned short* Wltx = Wlt;         // ldB = 2048, K = 1024
  const unsigned short* Wlth = Wlt + HH2;   // ldB = 2048, K = 1024

  // ---- 1) weights -> bf16, transposed ----
  cvt_transpose_kernel<<<dim3(HH1/32, EE/32),  256, 0, stream>>>(W1, W1t, EE,  HH1);
  cvt_transpose_kernel<<<dim3(HH2/32, HH1/32), 256, 0, stream>>>(W2, W2t, HH1, HH2);
  cvt_transpose_kernel<<<dim3(GN/32, (HH2+HHL)/32), 256, 0, stream>>>(Wl, Wlt, HH2+HHL, GN);

  // ---- 2) embedding gather ----
  embed_gather_kernel<<<NTOK, 256, 0, stream>>>(x, embed, Xemb);

  // ---- 3) MLP1: [16384,512]@[512,2048] + b1, ReLU -> bf16 ----
  gemm_bf16_kernel<128, true, true, false, false>
      <<<dim3(HH1/128, NTOK/128), 256, 0, stream>>>(Xemb, W1t, b1, nullptr,
                                                    H1a, NTOK, HH1, EE, EE);
  // ---- 4) MLP2: [16384,2048]@[2048,1024] + b2 -> bf16 ----
  gemm_bf16_kernel<128, false, true, false, false>
      <<<dim3(HH2/128, NTOK/128), 256, 0, stream>>>(H1a, W2t, b2, nullptr,
                                                    Hmlp, NTOK, HH2, HH1, HH1);
  // ---- 5) GX = Hmlp @ Wl_x + bl, stored time-major [t][b][4096] ----
  gemm_bf16_kernel<128, false, false, true, false>
      <<<dim3(GN/128, NTOK/128), 256, 0, stream>>>(Hmlp, Wltx, bl, nullptr,
                                                   GX, NTOK, GN, HH2, HH2 + HHL);
  // ---- 6) h0 = c0 = 0 ----
  lstm_init_kernel<<<(BB*HHL)/256, 256, 0, stream>>>(Cst, Hbf);

  // ---- 7) recurrence: gates = GX[t] + h_{t-1} @ Wl_h ; elementwise update ----
  for (int t = 0; t < TT; ++t) {
    gemm_bf16_kernel<64, false, false, false, true>
        <<<dim3(GN/128, 1), 256, 0, stream>>>(Hbf, Wlth, nullptr,
                                              GX + (size_t)t * BB * GN,
                                              GAT, BB, GN, HHL, HH2 + HHL);
    lstm_combine_kernel<<<(BB*HHL)/256, 256, 0, stream>>>(GAT, Cst, Hbf, out,
                                                          t == TT - 1 ? 1 : 0);
  }
}
